// MultiHeadAttention_22067541967461
// MI455X (gfx1250) — matrix operand order repaired and checked by reading, not compiled, not run on hardware
//
#include <hip/hip_runtime.h>
#include <stdint.h>

// ---------------------------------------------------------------------------
// MHA block for MI455X (gfx1250, wave32, WMMA).
// All GEMMs run on v_wmma_f32_16x16x32_bf16 (fp32 inputs converted to bf16 on
// the global->LDS path, fp32 accumulation). Attention-matrix traffic dominates
// (1 GiB attn_weights), so bias / 1/sqrt(dk) / residual are fused into GEMM
// epilogues and softmax + layernorm are single in-place streaming passes.
// ---------------------------------------------------------------------------

#define BATCH     4
#define SEQ       2048
#define DMODEL    1024
#define NHEADS    16
#define DK        64

typedef __bf16 bf16_t;
typedef __attribute__((ext_vector_type(16))) __bf16 v16bf;
typedef __attribute__((ext_vector_type(8)))  __bf16 v8bf;
typedef __attribute__((ext_vector_type(8)))  float  v8f;

// GEMM tiling: workgroup = 256 threads = 8 waves (wave32).
// Block tile 128x128x32; wave grid 2x4; each wave computes 64x32 = 4x2 WMMA tiles.
#define BM 128
#define BN 128
#define BK 32
#define RS (BK + 8)   // LDS row stride in bf16 elems: 80 bytes, 16B-aligned rows,
                      // bank-conflict-free ds_load_b128 fragment reads.

// --- WMMA fragment loads from LDS, per CDNA5 ISA §7.12.2 layouts ------------
// A 16x32 (MxK), 16-bit: lanes 0-15 -> M=lane, K = 0..7 & 16..23 ;
//                        lanes 16-31 -> M=lane-16, K = 8..15 & 24..31.
static __device__ __forceinline__ v16bf load_frag_A(const bf16_t* As, int rowBase) {
  const int lane = threadIdx.x & 31;
  const bf16_t* p = As + (rowBase + (lane & 15)) * RS + ((lane >> 4) * 8);
  v8bf lo = *(const v8bf*)p;           // K = kb .. kb+7
  v8bf hi = *(const v8bf*)(p + 16);    // K = kb+16 .. kb+23
  return __builtin_shufflevector(lo, hi, 0,1,2,3,4,5,6,7,8,9,10,11,12,13,14,15);
}
// B 32x16 (KxN), 16-bit: lanes 0-15 -> N=lane, K=0..7 & 16..23 ; lanes 16-31 -> K=8..15 & 24..31.
static __device__ __forceinline__ v16bf load_frag_B(const bf16_t* Bs, int colBase) {
  const int lane = threadIdx.x & 31;
  const bf16_t* p = Bs + (colBase + (lane & 15)) * RS + ((lane >> 4) * 8);
  v8bf lo = *(const v8bf*)p;           // K = kb .. kb+7
  v8bf hi = *(const v8bf*)(p + 16);    // K = kb+16 .. kb+23
  return __builtin_shufflevector(lo, hi, 0,1,2,3,4,5,6,7,8,9,10,11,12,13,14,15);
}

// ---------------------------------------------------------------------------
// Generic tiled GEMM:  C[m,n] = alpha * sum_k A[m,k] * B(k,n)  (+bias)(+res)
//   A: row-major [M,K], leading dim lda (k-contiguous).
//   TRANSB == false: B element (k,n) = B[n*ldbN + k]   (k-contiguous, e.g. W[N,K])
//   TRANSB == true : B element (k,n) = B[k*ldbK + n]   (n-contiguous, e.g. V[S,dk])
// Per-z (blockIdx.z) base offsets: zo = z/innerCnt, zi = z%innerCnt,
//   base += zo*str_o + zi*str_i  (handles [batch][head] sub-tensor views).
// M must be a multiple of 128 and K a multiple of 32 (true for every call);
// N is guarded (N=64 in the PV GEMM).
// ---------------------------------------------------------------------------
template <bool TRANSB>
__global__ __launch_bounds__(256)
void gemm_bf16_wmma(const float* __restrict__ A, int lda, long long sAo, long long sAi,
                    const float* __restrict__ B, int ldbN, int ldbK, long long sBo, long long sBi,
                    float* __restrict__ C, int ldc, long long sCo, long long sCi,
                    const float* __restrict__ bias,
                    const float* __restrict__ res, int ldres,
                    int M, int N, int K, float alpha, int innerCnt)
{
  __shared__ __align__(16) bf16_t As[BM * RS];
  __shared__ __align__(16) bf16_t Bs[BN * RS];

  const int z  = blockIdx.z;
  const int zo = z / innerCnt;
  const int zi = z % innerCnt;
  const float* Ab = A + zo * sAo + zi * sAi;
  const float* Bb = B + zo * sBo + zi * sBi;
  float*       Cb = C + zo * sCo + zi * sCi;

  const int m0 = blockIdx.x * BM;
  const int n0 = blockIdx.y * BN;

  const int tid  = threadIdx.x;
  const int wave = tid >> 5;
  const int wm   = wave & 1;   // 2 wave rows   -> 64 rows each
  const int wn   = wave >> 1;  // 4 wave cols   -> 32 cols each

  v8f acc[4][2];
#pragma unroll
  for (int mt = 0; mt < 4; ++mt)
#pragma unroll
    for (int nt = 0; nt < 2; ++nt)
#pragma unroll
      for (int i = 0; i < 8; ++i) acc[mt][nt][i] = 0.0f;

  for (int k0 = 0; k0 < K; k0 += BK) {
    // ---- stage A tile (128x32 fp32 -> bf16). 2 threads per row, 16 elems each.
    {
      const int r  = tid >> 1;
      const int c0 = (tid & 1) * 16;
      const float* src = Ab + (size_t)(m0 + r) * lda + (k0 + c0);
      if (k0 + BK < K) __builtin_prefetch(src + BK, 0, 1);   // global_prefetch next k-tile
      __align__(16) bf16_t tmp[16];
#pragma unroll
      for (int i = 0; i < 16; i += 4) {
        float4 f = *(const float4*)(src + i);
        tmp[i + 0] = (bf16_t)f.x; tmp[i + 1] = (bf16_t)f.y;
        tmp[i + 2] = (bf16_t)f.z; tmp[i + 3] = (bf16_t)f.w;
      }
      v8bf* dst = (v8bf*)(As + r * RS + c0);
      dst[0] = *(const v8bf*)(tmp);
      dst[1] = *(const v8bf*)(tmp + 8);
    }
    // ---- stage B tile (128 cols x 32 k) as Bs[n][k], k-major bf16.
    if (!TRANSB) {
      const int r  = tid >> 1;
      const int c0 = (tid & 1) * 16;
      __align__(16) bf16_t tmp[16];
      if (n0 + r < N) {
        const float* src = Bb + (size_t)(n0 + r) * ldbN + (k0 + c0);
        if (k0 + BK < K) __builtin_prefetch(src + BK, 0, 1);
#pragma unroll
        for (int i = 0; i < 16; i += 4) {
          float4 f = *(const float4*)(src + i);
          tmp[i + 0] = (bf16_t)f.x; tmp[i + 1] = (bf16_t)f.y;
          tmp[i + 2] = (bf16_t)f.z; tmp[i + 3] = (bf16_t)f.w;
        }
      } else {
#pragma unroll
        for (int i = 0; i < 16; ++i) tmp[i] = (bf16_t)0.0f;
      }
      v8bf* dst = (v8bf*)(Bs + r * RS + c0);
      dst[0] = *(const v8bf*)(tmp);
      dst[1] = *(const v8bf*)(tmp + 8);
    } else {
      // B is n-contiguous per k (e.g. V): threads sweep n for coalescing,
      // transpose into k-major LDS with scalar b16 stores.
      const int kk = tid >> 3;          // 0..31
      const int ns = (tid & 7) * 16;    // 0,16,..,112
#pragma unroll
      for (int j = 0; j < 16; ++j) {
        const int n = ns + j;
        float f = 0.0f;
        if (n0 + n < N)
          f = Bb[(size_t)(k0 + kk) * ldbK + (size_t)(n0 + n) * ldbN];
        Bs[n * RS + kk] = (bf16_t)f;
      }
    }
    __syncthreads();

    // ---- fragments + 8 WMMAs per wave (BK == WMMA K == 32).
    v16bf af[4];
#pragma unroll
    for (int mt = 0; mt < 4; ++mt) af[mt] = load_frag_A(As, wm * 64 + mt * 16);
    v16bf bfrag[2];
#pragma unroll
    for (int nt = 0; nt < 2; ++nt) bfrag[nt] = load_frag_B(Bs, wn * 32 + nt * 16);
#pragma unroll
    for (int mt = 0; mt < 4; ++mt)
#pragma unroll
      for (int nt = 0; nt < 2; ++nt)
        acc[mt][nt] = __builtin_amdgcn_wmma_f32_16x16x32_bf16(
            false, af[mt], false, bfrag[nt], (short)0, acc[mt][nt], false, false);
    __syncthreads();
  }

  // ---- epilogue: C/D layout is VGPR i -> M = i (lanes 0-15) / 8+i (lanes 16-31),
  //                N = lane&15.
  const int lane = tid & 31;
  const int nl   = lane & 15;
  const int mh   = (lane >> 4) * 8;
#pragma unroll
  for (int mt = 0; mt < 4; ++mt) {
#pragma unroll
    for (int nt = 0; nt < 2; ++nt) {
      const int n = n0 + wn * 32 + nt * 16 + nl;
      if (n < N) {
        const float bv = bias ? bias[n] : 0.0f;
#pragma unroll
        for (int i = 0; i < 8; ++i) {
          const int m = m0 + wm * 64 + mt * 16 + mh + i;
          float v = alpha * acc[mt][nt][i] + bv;
          if (res) v += res[(size_t)m * ldres + n];
          Cb[(size_t)m * ldc + n] = v;
        }
      }
    }
  }
}

// ---------------------------------------------------------------------------
// In-place row softmax over 2048 columns. One 256-thread block per row.
// ---------------------------------------------------------------------------
__global__ __launch_bounds__(256)
void softmax2048(float* __restrict__ W)
{
  float* row = W + (size_t)blockIdx.x * SEQ;
  const int t = threadIdx.x;
  float4 a = *(const float4*)(row + t * 8);
  float4 b = *(const float4*)(row + t * 8 + 4);

  float mx = fmaxf(fmaxf(fmaxf(a.x, a.y), fmaxf(a.z, a.w)),
                   fmaxf(fmaxf(b.x, b.y), fmaxf(b.z, b.w)));
#pragma unroll
  for (int off = 16; off > 0; off >>= 1) mx = fmaxf(mx, __shfl_xor(mx, off, 32));

  __shared__ float smax[8];
  __shared__ float ssum[8];
  if ((t & 31) == 0) smax[t >> 5] = mx;
  __syncthreads();
  mx = smax[0];
#pragma unroll
  for (int i = 1; i < 8; ++i) mx = fmaxf(mx, smax[i]);

  float e[8];
  e[0] = __expf(a.x - mx); e[1] = __expf(a.y - mx);
  e[2] = __expf(a.z - mx); e[3] = __expf(a.w - mx);
  e[4] = __expf(b.x - mx); e[5] = __expf(b.y - mx);
  e[6] = __expf(b.z - mx); e[7] = __expf(b.w - mx);
  float s = ((e[0] + e[1]) + (e[2] + e[3])) + ((e[4] + e[5]) + (e[6] + e[7]));
#pragma unroll
  for (int off = 16; off > 0; off >>= 1) s += __shfl_xor(s, off, 32);
  if ((t & 31) == 0) ssum[t >> 5] = s;
  __syncthreads();
  s = 0.0f;
#pragma unroll
  for (int i = 0; i < 8; ++i) s += ssum[i];
  const float inv = 1.0f / s;

  a.x = e[0] * inv; a.y = e[1] * inv; a.z = e[2] * inv; a.w = e[3] * inv;
  b.x = e[4] * inv; b.y = e[5] * inv; b.z = e[6] * inv; b.w = e[7] * inv;
  *(float4*)(row + t * 8)     = a;
  *(float4*)(row + t * 8 + 4) = b;
}

// ---------------------------------------------------------------------------
// In-place LayerNorm over 1024 columns (biased variance, eps=1e-5).
// One 256-thread block per row; each thread owns 4 contiguous columns.
// ---------------------------------------------------------------------------
__global__ __launch_bounds__(256)
void layernorm1024(float* __restrict__ Y,
                   const float* __restrict__ gamma,
                   const float* __restrict__ beta)
{
  float* row = Y + (size_t)blockIdx.x * DMODEL;
  const int t = threadIdx.x;
  float4 x = *(const float4*)(row + t * 4);

  float s1 = (x.x + x.y) + (x.z + x.w);
  float s2 = (x.x * x.x + x.y * x.y) + (x.z * x.z + x.w * x.w);
#pragma unroll
  for (int off = 16; off > 0; off >>= 1) {
    s1 += __shfl_xor(s1, off, 32);
    s2 += __shfl_xor(s2, off, 32);
  }
  __shared__ float r1[8];
  __shared__ float r2[8];
  if ((t & 31) == 0) { r1[t >> 5] = s1; r2[t >> 5] = s2; }
  __syncthreads();
  s1 = 0.0f; s2 = 0.0f;
#pragma unroll
  for (int i = 0; i < 8; ++i) { s1 += r1[i]; s2 += r2[i]; }

  const float mu   = s1 * (1.0f / DMODEL);
  const float var  = s2 * (1.0f / DMODEL) - mu * mu;
  const float rstd = rsqrtf(var + 1e-5f);

  const float4 g = *(const float4*)(gamma + t * 4);
  const float4 bb = *(const float4*)(beta + t * 4);
  x.x = (x.x - mu) * rstd * g.x + bb.x;
  x.y = (x.y - mu) * rstd * g.y + bb.y;
  x.z = (x.z - mu) * rstd * g.z + bb.z;
  x.w = (x.w - mu) * rstd * g.w + bb.w;
  *(float4*)(row + t * 4) = x;
}

// ---------------------------------------------------------------------------
extern "C" void kernel_launch(void* const* d_in, const int* in_sizes, int n_in,
                              void* d_out, int out_size, void* d_ws, size_t ws_size,
                              hipStream_t stream)
{
  const float* q    = (const float*)d_in[0];
  const float* k    = (const float*)d_in[1];
  const float* v    = (const float*)d_in[2];
  const float* wq_w = (const float*)d_in[3];
  const float* wq_b = (const float*)d_in[4];
  const float* wk_w = (const float*)d_in[5];
  const float* wk_b = (const float*)d_in[6];
  const float* wv_w = (const float*)d_in[7];
  const float* wv_b = (const float*)d_in[8];
  const float* wo_w = (const float*)d_in[9];
  const float* wo_b = (const float*)d_in[10];
  const float* ln_g = (const float*)d_in[11];
  const float* ln_b = (const float*)d_in[12];

  const long long P  = (long long)BATCH * SEQ * DMODEL;  // 8,388,608 elems
  const long long SS = (long long)SEQ * SEQ;             // 4,194,304 elems
  const long long SD = (long long)SEQ * DMODEL;

  float* Qp = (float*)d_ws;      // [B*S, D] projected Q
  float* Kp = Qp + P;            // [B*S, D] projected K
  float* Vp = Kp + P;            // [B*S, D] projected V
  float* AO = Qp;                // attention output reuses Qp (dead after QK^T)

  float* outp  = (float*)d_out;  // [B, S, D]    tuple element 0
  float* attnW = outp + P;       // [B, H, S, S] tuple element 1

  const dim3 blk(256);

  // --- Q/K/V projections: [8192,1024] x W[1024,1024]^T + bias ------------
  const dim3 gProj(BATCH * SEQ / BM, DMODEL / BN, 1);
  gemm_bf16_wmma<false><<<gProj, blk, 0, stream>>>(
      q, DMODEL, 0, 0,  wq_w, DMODEL, 1, 0, 0,
      Qp, DMODEL, 0, 0, wq_b, nullptr, 0,
      BATCH * SEQ, DMODEL, DMODEL, 1.0f, NHEADS);
  gemm_bf16_wmma<false><<<gProj, blk, 0, stream>>>(
      k, DMODEL, 0, 0,  wk_w, DMODEL, 1, 0, 0,
      Kp, DMODEL, 0, 0, wk_b, nullptr, 0,
      BATCH * SEQ, DMODEL, DMODEL, 1.0f, NHEADS);
  gemm_bf16_wmma<false><<<gProj, blk, 0, stream>>>(
      v, DMODEL, 0, 0,  wv_w, DMODEL, 1, 0, 0,
      Vp, DMODEL, 0, 0, wv_b, nullptr, 0,
      BATCH * SEQ, DMODEL, DMODEL, 1.0f, NHEADS);

  // --- scores = Qh . Kh^T / sqrt(dk), per (b,h), written into attn_weights
  const dim3 gScore(SEQ / BM, SEQ / BN, BATCH * NHEADS);
  gemm_bf16_wmma<false><<<gScore, blk, 0, stream>>>(
      Qp, DMODEL, SD, DK,
      Kp, DMODEL, 1, SD, DK,
      attnW, SEQ, (long long)NHEADS * SS, SS,
      nullptr, nullptr, 0,
      SEQ, SEQ, DK, 0.125f, NHEADS);

  // --- softmax in place on attn_weights ----------------------------------
  softmax2048<<<dim3(BATCH * NHEADS * SEQ), blk, 0, stream>>>(attnW);

  // --- attn_out = P . Vh, per (b,h): M=2048, N=64, K=2048 (B transposed) --
  const dim3 gAV(SEQ / BM, 1, BATCH * NHEADS);
  gemm_bf16_wmma<true><<<gAV, blk, 0, stream>>>(
      attnW, SEQ, (long long)NHEADS * SS, SS,
      Vp, 1, DMODEL, SD, DK,
      AO, DMODEL, SD, DK,            // writes cols [h*64, h*64+64) -> combined [B,S,D]
      nullptr, nullptr, 0,
      SEQ, DK, SEQ, 1.0f, NHEADS);

  // --- out = AO . Wo^T + bo + residual(q) --------------------------------
  gemm_bf16_wmma<false><<<gProj, blk, 0, stream>>>(
      AO, DMODEL, 0, 0,  wo_w, DMODEL, 1, 0, 0,
      outp, DMODEL, 0, 0, wo_b, q, DMODEL,
      BATCH * SEQ, DMODEL, DMODEL, 1.0f, NHEADS);

  // --- layernorm in place on out -----------------------------------------
  layernorm1024<<<dim3(BATCH * SEQ), blk, 0, stream>>>(outp, ln_g, ln_b);
}
